// CrossAttUnit_49770081026328
// MI455X (gfx1250) — matrix operand order repaired and checked by reading, not compiled, not run on hardware
//
#include <hip/hip_runtime.h>
#include <math.h>

// CDNA5 / gfx1250 wave32 WMMA types
typedef __attribute__((ext_vector_type(16))) __bf16 v16bf;
typedef __attribute__((ext_vector_type(8)))  __bf16 v8bf;
typedef __attribute__((ext_vector_type(8)))  float  v8f;

#define Hdim 256   // pc_hidden
#define Ddim 64    // cross_att_dim
#define Ldim 64    // seg_len
#define YK_PAD 80  // *2B = 160 bytes (32*5): every vector access >=16B aligned
#define M_PAD  65  // f32 rows, bank-conflict-free row & column walks

// Build the 16-bit A-matrix operand (16x32, MxK) from two 8-float chunks,
// converting f32 -> bf16. Element e of v16bf lives in VGPR e/2; the caller
// passes contiguous K (k = 16h + e, same as B), not the ISA table order.
__device__ inline v16bf make_a_f32(float4 a0, float4 a1, float4 b0, float4 b1) {
    v16bf A;
    A[0]  = (__bf16)a0.x; A[1]  = (__bf16)a0.y; A[2]  = (__bf16)a0.z; A[3]  = (__bf16)a0.w;
    A[4]  = (__bf16)a1.x; A[5]  = (__bf16)a1.y; A[6]  = (__bf16)a1.z; A[7]  = (__bf16)a1.w;
    A[8]  = (__bf16)b0.x; A[9]  = (__bf16)b0.y; A[10] = (__bf16)b0.z; A[11] = (__bf16)b0.w;
    A[12] = (__bf16)b1.x; A[13] = (__bf16)b1.y; A[14] = (__bf16)b1.z; A[15] = (__bf16)b1.w;
    return A;
}

// 16 contiguous bf16 (>=16B aligned, global or LDS) -> B-matrix operand.
__device__ inline v16bf load_b16(const __bf16* p) {
    v8bf lo = *(const v8bf*)p;
    v8bf hi = *(const v8bf*)(p + 8);
    v16bf r;
#pragma unroll
    for (int i = 0; i < 8; ++i) { r[i] = lo[i]; r[i + 8] = hi[i]; }
    return r;
}

// bf16 A operand from two 16B LDS chunks (second-GEMM A loads).
__device__ inline v16bf make_a_lds(const __bf16* p) {
    v8bf lo = *(const v8bf*)p;
    v8bf hi = *(const v8bf*)(p + 8);
    v16bf r;
#pragma unroll
    for (int i = 0; i < 8; ++i) { r[i] = lo[i]; r[i + 8] = hi[i]; }
    return r;
}

// ---- one-time prep: kT/qT = transpose(k/q) as bf16 in d_ws -----------------
// kT[c*256 + r] = k[r*64 + c].  Output-driven so stores are coalesced;
// the 64 KB result lives in L2 (192 MB) for the whole main dispatch.
__global__ __launch_bounds__(256) void prep_kq_kernel(
    const float* __restrict__ kmat, const float* __restrict__ qmat,
    __bf16* __restrict__ kT, __bf16* __restrict__ qT)
{
    const int e = blockIdx.x * 256 + threadIdx.x;   // 0..16383
    const int c = e >> 8;                           // d index      0..63
    const int r = e & 255;                          // hidden index 0..255
    kT[e] = (__bf16)kmat[r * Ddim + c];
    qT[e] = (__bf16)qmat[r * Ddim + c];
}

__global__ __launch_bounds__(128) void cross_att_unit_kernel(
    const float* __restrict__ yhat,   // [N, 256]
    const float* __restrict__ y,      // [N, 256]
    const __bf16* __restrict__ kT,    // [64, 256] = k^T, bf16
    const __bf16* __restrict__ qT,    // [64, 256] = q^T, bf16
    float* __restrict__ out)          // [B, 64, 64]
{
    __shared__ __bf16 sYK[Ldim][YK_PAD];   // y @ k   (bf16, row-major)
    __shared__ __bf16 sYQ[Ldim][YK_PAD];   // yhat @ q
    __shared__ float  sM [Ldim][M_PAD];    // scores -> attn
    __shared__ float  sCol[Ldim];

    const int tid  = threadIdx.x;
    const int lane = tid & 31;
    const int w    = tid >> 5;             // wave 0..3, owns rows [16w,16w+16)
    const int seg  = blockIdx.x;

    // Lane -> matrix mapping for 16-bit WMMA (wave32; rows/cols per ISA, K contiguous):
    const int col16 = lane & 15;
    const int hs_a  = (lane < 16) ? 0 : 8;   // unused: A loads use hs_b like B
    const int hs_b  = (lane < 16) ? 0 : 16;  // B: K offset of this lane's half
    const int mrow  = (w << 4) + col16;      // A-operand row this lane feeds
    const int srow  = (w << 4) + ((lane < 16) ? 0 : 8); // C/D row base

    // ---- projection GEMM: dst[64x64] = src_seg[64x256] @ BT^T --------------
    // B-operand comes straight from global (L2-resident bf16 k^T/q^T):
    // 16 contiguous bf16 per lane per (kc, ntile).
    auto project = [&](const float* __restrict__ src,
                       const __bf16* __restrict__ BT,
                       __bf16 (*dst)[YK_PAD]) {
        v8f acc[4] = {v8f{}, v8f{}, v8f{}, v8f{}};
        const float*  row   = src + ((size_t)seg * Ldim + mrow) * Hdim;
        const __bf16* bbase = BT + (size_t)col16 * Hdim + hs_b;
#pragma unroll
        for (int kc = 0; kc < Hdim; kc += 32) {
            const float4* p0 = (const float4*)(row + kc + hs_b);
            const float4* p1 = (const float4*)(row + kc + hs_b + 8);
            const v16bf A = make_a_f32(p0[0], p0[1], p1[0], p1[1]);
            // Preload all 4 N-tile B operands so the loads pipeline,
            // then issue the 4 WMMAs back-to-back.
            v16bf B[4];
#pragma unroll
            for (int nt = 0; nt < 4; ++nt)
                B[nt] = load_b16(bbase + (size_t)nt * 16 * Hdim + kc);
#pragma unroll
            for (int nt = 0; nt < 4; ++nt)
                acc[nt] = __builtin_amdgcn_wmma_f32_16x16x32_bf16(
                    false, A, false, B[nt], (short)0, acc[nt], false, false);
        }
        // C/D layout: VGPR r, lanes 0-15 -> row srow+r, lanes 16-31 -> +8
#pragma unroll
        for (int nt = 0; nt < 4; ++nt)
#pragma unroll
            for (int r = 0; r < 8; ++r)
                dst[srow + r][nt * 16 + col16] = (__bf16)acc[nt][r];
    };

    project(y,    kT, sYK);   // yk = y_seg    @ k
    project(yhat, qT, sYQ);   // yq = yhat_seg @ q
    __syncthreads();

    // ---- scores: M = (1/8) * YK @ YQ^T  (K = 64, two 32-steps) -------------
    {
        v8f macc[4] = {v8f{}, v8f{}, v8f{}, v8f{}};
#pragma unroll
        for (int dc = 0; dc < Ddim; dc += 32) {
            const v16bf A = make_a_lds(&sYK[mrow][dc + hs_b]);
            v16bf B[4];
#pragma unroll
            for (int nt = 0; nt < 4; ++nt)
                B[nt] = load_b16(&sYQ[nt * 16 + col16][dc + hs_b]);
#pragma unroll
            for (int nt = 0; nt < 4; ++nt)
                macc[nt] = __builtin_amdgcn_wmma_f32_16x16x32_bf16(
                    false, A, false, B[nt], (short)0, macc[nt], false, false);
        }
        const float scale = 0.125f;   // 1/sqrt(64)
#pragma unroll
        for (int nt = 0; nt < 4; ++nt)
#pragma unroll
            for (int r = 0; r < 8; ++r)
                sM[srow + r][nt * 16 + col16] = scale * macc[nt][r];
    }
    __syncthreads();

    // ---- softmax over axis=2 (row-wise), + EPS -----------------------------
    if (tid < Ldim) {
        float mx = -INFINITY;
#pragma unroll 8
        for (int m = 0; m < Ldim; ++m) mx = fmaxf(mx, sM[tid][m]);
        float s = 0.f;
#pragma unroll 8
        for (int m = 0; m < Ldim; ++m) {
            const float e = __expf(sM[tid][m] - mx);
            sM[tid][m] = e;
            s += e;
        }
        const float inv = 1.f / s;
#pragma unroll 8
        for (int m = 0; m < Ldim; ++m) sM[tid][m] = sM[tid][m] * inv + 1e-6f;
    }
    __syncthreads();

    // ---- column sums (axis=1 normalization) --------------------------------
    if (tid < Ldim) {
        float s = 0.f;
#pragma unroll 8
        for (int l = 0; l < Ldim; ++l) s += sM[l][tid];
        sCol[tid] = 1.f / s;
    }
    __syncthreads();

    // ---- coalesced store of W[seg] -----------------------------------------
    float* o = out + (size_t)seg * (Ldim * Ldim);
#pragma unroll
    for (int i = 0; i < (Ldim * Ldim) / 128; ++i) {
        const int e = i * 128 + tid;
        o[e] = sM[e >> 6][e & 63] * sCol[e & 63];
    }
}

extern "C" void kernel_launch(void* const* d_in, const int* in_sizes, int n_in,
                              void* d_out, int out_size, void* d_ws, size_t ws_size,
                              hipStream_t stream) {
    const float* yhat = (const float*)d_in[0];
    const float* y    = (const float*)d_in[1];
    const float* kmat = (const float*)d_in[2];
    const float* qmat = (const float*)d_in[3];
    float* out = (float*)d_out;

    // Workspace layout: kT bf16 [64*256], qT bf16 [64*256]  (64 KB total)
    __bf16* kT = (__bf16*)d_ws;
    __bf16* qT = kT + Hdim * Ddim;

    const int N = in_sizes[0] / Hdim;   // total tokens (262144)
    const int B = N / Ldim;             // segments (4096)

    prep_kq_kernel<<<dim3((Hdim * Ddim) / 256), dim3(256), 0, stream>>>(kmat, qmat, kT, qT);
    cross_att_unit_kernel<<<dim3(B), dim3(128), 0, stream>>>(yhat, y, kT, qT, out);
}
